// RGCNLayer_31001073943194
// MI455X (gfx1250) — matrix operand order repaired and checked by reading, not compiled, not run on hardware
//
#include <hip/hip_runtime.h>

#define N_NODES 100000
#define N_REL   8
#define N_EDGE  200000
#define D_IN    128
#define D_EDGE  32
#define D_K     160   // D_IN + D_EDGE
#define D_HID   128
#define LN_EPS  1e-6f

typedef _Float16 v16h __attribute__((ext_vector_type(16)));
typedef _Float16 h8v  __attribute__((ext_vector_type(8)));
typedef float    v8f  __attribute__((ext_vector_type(8)));

__device__ __forceinline__ v16h combine(h8v lo, h8v hi) {
  v16h r;
#pragma unroll
  for (int i = 0; i < 8; i++) { r[i] = lo[i]; r[i + 8] = hi[i]; }
  return r;
}

// Non-returning, agent-scope fp32 atomic add -> native global_atomic_add_f32
__device__ __forceinline__ void atomic_add_f32(float* p, float v) {
  (void)__hip_atomic_fetch_add(p, v, __ATOMIC_RELAXED, __HIP_MEMORY_SCOPE_AGENT);
}

// ---------------------------------------------------------------- utilities
__global__ void zero_kernel(float* __restrict__ out, int* __restrict__ deg) {
  const size_t stride = (size_t)gridDim.x * blockDim.x;
  const size_t base = (size_t)blockIdx.x * blockDim.x + threadIdx.x;
  for (size_t i = base; i < (size_t)N_NODES * D_HID; i += stride) out[i] = 0.0f;
  for (size_t i = base; i < (size_t)N_REL * N_NODES; i += stride) deg[i] = 0;
}

__global__ void degree_kernel(const int* __restrict__ receivers, int* __restrict__ deg) {
  const size_t stride = (size_t)gridDim.x * blockDim.x;
  const size_t base = (size_t)blockIdx.x * blockDim.x + threadIdx.x;
  for (size_t i = base; i < (size_t)N_REL * N_EDGE; i += stride) {
    const int r = (int)(i / N_EDGE);
    atomicAdd(&deg[(size_t)r * N_NODES + receivers[i]], 1);
  }
}

__global__ void invdeg_kernel(int* __restrict__ deg) {
  float* fd = (float*)deg;  // in-place int -> float (per-thread, kernel-ordered)
  const size_t stride = (size_t)gridDim.x * blockDim.x;
  const size_t base = (size_t)blockIdx.x * blockDim.x + threadIdx.x;
  for (size_t i = base; i < (size_t)N_REL * N_NODES; i += stride) {
    const int d = deg[i];
    fd[i] = 1.0f / (float)(d < 1 ? 1 : d);
  }
}

// ---------------------------------------------------------------- edge GEMM
#define WP  168   // W LDS pitch (halves): 336B rows, 16B aligned, conflict-free b128
#define FP  160   // feat LDS pitch (halves)
#define TM  64    // edges per tile
#define BPR 256   // persistent blocks per relation

__global__ __launch_bounds__(256) void edge_gemm_kernel(
    const float* __restrict__ nodes, const float* __restrict__ edges,
    const int* __restrict__ senders, const int* __restrict__ receivers,
    const float* __restrict__ W_rel, const float* __restrict__ invdeg,
    float* __restrict__ agg)
{
  __shared__ __align__(16) _Float16 sW[D_HID * WP];  // W_rel[r] transposed: [n][k]
  __shared__ __align__(16) _Float16 sF[TM * FP];     // gathered features [m][k]
  __shared__ int   sRcv[TM];
  __shared__ float sInv[TM];

  const int tid  = threadIdx.x;
  const int lane = tid & 31;
  const int wave = tid >> 5;
  const int r    = blockIdx.x / BPR;
  const int bt   = blockIdx.x % BPR;

  // stage W_rel[r] (row-major [K=160][N=128]) into LDS as sW[n*WP + k], f16
  const float* Wr = W_rel + (size_t)r * D_K * D_HID;
  for (int idx = tid; idx < D_K * D_HID; idx += 256) {
    const int k = idx >> 7;
    const int n = idx & 127;
    sW[n * WP + k] = (_Float16)Wr[idx];
  }
  __syncthreads();

  const int mb  = wave >> 1;            // 16-row band 0..3
  const int nh  = wave & 1;             // 64-col half 0..1
  const int mlo = (lane < 16) ? 0 : 8;  // C: VGPR j -> row j or j+8
  const int ncl = lane & 15;            // C/B column within 16-tile
  const int am  = lane & 15;            // A row within band
  const int alo = (lane < 16) ? 0 : 8;  // A K sub-offset (ISA 7.12.2 16-bit A layout)
  const int blo = (lane < 16) ? 0 : 8;  // B K sub-offset

  const int tiles = N_EDGE / TM;        // 3125 exact
  for (int t = bt; t < tiles; t += BPR) {
    const int e0 = t * TM;

    // ---- gather: 4 threads per edge row ----
    {
      const int g   = tid >> 2;
      const int sub = tid & 3;
      const int e   = e0 + g;
      if (sub == 0) {
        const int rcv = (e < N_EDGE) ? receivers[(size_t)r * N_EDGE + e] : -1;
        sRcv[g] = rcv;
        sInv[g] = (rcv >= 0) ? invdeg[(size_t)r * N_NODES + rcv] : 0.0f;
      }
      const float4* nsrc = nullptr;
      const float4* esrc = nullptr;
      if (e < N_EDGE) {
        const int s = senders[(size_t)r * N_EDGE + e];
        nsrc = (const float4*)(nodes + (size_t)s * D_IN) + sub * 8;
        esrc = (const float4*)(edges + ((size_t)r * N_EDGE + e) * D_EDGE) + sub * 2;
      }
      _Float16* dn = sF + g * FP + sub * 32;
#pragma unroll
      for (int j = 0; j < 8; j++) {
        const float4 v = nsrc ? nsrc[j] : make_float4(0.f, 0.f, 0.f, 0.f);
        dn[j*4+0] = (_Float16)v.x; dn[j*4+1] = (_Float16)v.y;
        dn[j*4+2] = (_Float16)v.z; dn[j*4+3] = (_Float16)v.w;
      }
      _Float16* de = sF + g * FP + D_IN + sub * 8;
#pragma unroll
      for (int j = 0; j < 2; j++) {
        const float4 v = esrc ? esrc[j] : make_float4(0.f, 0.f, 0.f, 0.f);
        de[j*4+0] = (_Float16)v.x; de[j*4+1] = (_Float16)v.y;
        de[j*4+2] = (_Float16)v.z; de[j*4+3] = (_Float16)v.w;
      }
    }
    __syncthreads();

    // ---- WMMA: wave computes its 16x64 slice of the 64x128 tile ----
    v8f c0 = {}, c1 = {}, c2 = {}, c3 = {};
#pragma unroll
    for (int kc = 0; kc < 5; kc++) {
      const int kb = kc * 32;
      const _Float16* ap = sF + (mb * 16 + am) * FP + kb + alo;
      const v16h a = combine(*(const h8v*)ap, *(const h8v*)(ap + 16));
      const _Float16* bp0 = sW + (nh * 64 +  0 + ncl) * WP + kb + blo;
      const _Float16* bp1 = sW + (nh * 64 + 16 + ncl) * WP + kb + blo;
      const _Float16* bp2 = sW + (nh * 64 + 32 + ncl) * WP + kb + blo;
      const _Float16* bp3 = sW + (nh * 64 + 48 + ncl) * WP + kb + blo;
      const v16h b0 = combine(*(const h8v*)bp0, *(const h8v*)(bp0 + 16));
      const v16h b1 = combine(*(const h8v*)bp1, *(const h8v*)(bp1 + 16));
      const v16h b2 = combine(*(const h8v*)bp2, *(const h8v*)(bp2 + 16));
      const v16h b3 = combine(*(const h8v*)bp3, *(const h8v*)(bp3 + 16));
      c0 = __builtin_amdgcn_wmma_f32_16x16x32_f16(false, a, false, b0, (short)0, c0, false, false);
      c1 = __builtin_amdgcn_wmma_f32_16x16x32_f16(false, a, false, b1, (short)0, c1, false, false);
      c2 = __builtin_amdgcn_wmma_f32_16x16x32_f16(false, a, false, b2, (short)0, c2, false, false);
      c3 = __builtin_amdgcn_wmma_f32_16x16x32_f16(false, a, false, b3, (short)0, c3, false, false);
    }

    // ---- degree-scaled scatter-add into aggregate (native fp32 atomics) ----
#pragma unroll
    for (int j = 0; j < 8; j++) {
      const int m   = mb * 16 + mlo + j;
      const int rcv = sRcv[m];
      if (rcv >= 0) {
        const float inv = sInv[m];
        float* dst = agg + (size_t)rcv * D_HID + nh * 64 + ncl;
        atomic_add_f32(dst +  0, c0[j] * inv);
        atomic_add_f32(dst + 16, c1[j] * inv);
        atomic_add_f32(dst + 32, c2[j] * inv);
        atomic_add_f32(dst + 48, c3[j] * inv);
      }
    }
    __syncthreads();  // protect sF/sRcv/sInv reuse next tile
  }
}

// ------------------------------------------------- node proj + LN + ReLU
#define WP2 136   // 272B rows: 16B aligned, conflict-free
#define FP2 136
#define SHP 132   // h tile pitch (floats)
#define TM2 32    // nodes per tile (100000 / 32 = 3125 exact)

__global__ __launch_bounds__(256) void final_kernel(
    const float* __restrict__ nodes, const float* __restrict__ W_node,
    const float* __restrict__ ln_scale, const float* __restrict__ ln_bias,
    float* __restrict__ out)
{
  __shared__ __align__(16) _Float16 sW[D_HID * WP2];
  __shared__ __align__(16) unsigned char uBuf[TM2 * SHP * 4]; // sF (f16) then sH (f32)
  __shared__ float ps[TM2 * 8], pq[TM2 * 8];
  __shared__ float smu[TM2], srs[TM2];

  _Float16* sF = (_Float16*)uBuf;
  float*    sH = (float*)uBuf;

  const int tid  = threadIdx.x;
  const int lane = tid & 31;
  const int wave = tid >> 5;
  const int n0   = blockIdx.x * TM2;

  for (int idx = tid; idx < D_IN * D_HID; idx += 256) {
    const int k = idx >> 7, n = idx & 127;
    sW[n * WP2 + k] = (_Float16)W_node[idx];
  }
  {
    const int g = tid >> 3, sub = tid & 7;
    const int row = n0 + g;
    const float4* src = (row < N_NODES)
        ? (const float4*)(nodes + (size_t)row * D_IN) + sub * 4 : nullptr;
    _Float16* dst = sF + g * FP2 + sub * 16;
#pragma unroll
    for (int j = 0; j < 4; j++) {
      const float4 v = src ? src[j] : make_float4(0.f, 0.f, 0.f, 0.f);
      dst[j*4+0] = (_Float16)v.x; dst[j*4+1] = (_Float16)v.y;
      dst[j*4+2] = (_Float16)v.z; dst[j*4+3] = (_Float16)v.w;
    }
  }
  __syncthreads();

  const int mb  = wave >> 2;            // 0..1
  const int nq  = wave & 3;             // 0..3 (two 16-col tiles each)
  const int mlo = (lane < 16) ? 0 : 8;
  const int ncl = lane & 15;
  const int am  = lane & 15;
  const int alo = (lane < 16) ? 0 : 8;
  const int blo = (lane < 16) ? 0 : 8;

  v8f c0 = {}, c1 = {};
#pragma unroll
  for (int kc = 0; kc < 4; kc++) {
    const int kb = kc * 32;
    const _Float16* ap = sF + (mb * 16 + am) * FP2 + kb + alo;
    const v16h a = combine(*(const h8v*)ap, *(const h8v*)(ap + 16));
    const _Float16* bp0 = sW + ((nq * 2 + 0) * 16 + ncl) * WP2 + kb + blo;
    const _Float16* bp1 = sW + ((nq * 2 + 1) * 16 + ncl) * WP2 + kb + blo;
    const v16h b0 = combine(*(const h8v*)bp0, *(const h8v*)(bp0 + 16));
    const v16h b1 = combine(*(const h8v*)bp1, *(const h8v*)(bp1 + 16));
    c0 = __builtin_amdgcn_wmma_f32_16x16x32_f16(false, a, false, b0, (short)0, c0, false, false);
    c1 = __builtin_amdgcn_wmma_f32_16x16x32_f16(false, a, false, b1, (short)0, c1, false, false);
  }
  __syncthreads();  // sF dead; reuse buffer as sH

#pragma unroll
  for (int j = 0; j < 8; j++) {
    const int m = mb * 16 + mlo + j;
    sH[m * SHP + (nq * 2 + 0) * 16 + ncl] = c0[j];
    sH[m * SHP + (nq * 2 + 1) * 16 + ncl] = c1[j];
  }
  __syncthreads();

  // h += aggregate (accumulated in `out` by edge kernel)
  for (int idx = tid; idx < TM2 * D_HID; idx += 256) {
    const int m = idx >> 7, cidx = idx & 127;
    const int row = n0 + m;
    if (row < N_NODES) sH[m * SHP + cidx] += out[(size_t)row * D_HID + cidx];
  }
  __syncthreads();

  // LayerNorm partials: 8 threads per row, 16 cols each
  {
    const int m = tid >> 3, sub = tid & 7;
    float s = 0.f, q = 0.f;
    const float* hr = sH + m * SHP + sub * 16;
#pragma unroll
    for (int j = 0; j < 16; j++) { const float v = hr[j]; s += v; q += v * v; }
    ps[m * 8 + sub] = s; pq[m * 8 + sub] = q;
  }
  __syncthreads();
  if (tid < TM2) {
    float s = 0.f, q = 0.f;
#pragma unroll
    for (int j = 0; j < 8; j++) { s += ps[tid * 8 + j]; q += pq[tid * 8 + j]; }
    const float mu = s * (1.0f / D_HID);
    float var = q * (1.0f / D_HID) - mu * mu;
    var = var < 0.f ? 0.f : var;
    smu[tid] = mu;
    srs[tid] = __frsqrt_rn(var + LN_EPS);
  }
  __syncthreads();
  {
    const int m = tid >> 3, sub = tid & 7;
    const int row = n0 + m;
    if (row < N_NODES) {
      const float mu = smu[m], rs = srs[m];
      const float* hr = sH + m * SHP + sub * 16;
      float* dst = out + (size_t)row * D_HID + sub * 16;
#pragma unroll
      for (int j = 0; j < 16; j++) {
        const int cidx = sub * 16 + j;
        const float v = (hr[j] - mu) * rs * ln_scale[cidx] + ln_bias[cidx];
        dst[j] = v > 0.f ? v : 0.f;
      }
    }
  }
}

// ---------------------------------------------------------------- launcher
extern "C" void kernel_launch(void* const* d_in, const int* in_sizes, int n_in,
                              void* d_out, int out_size, void* d_ws, size_t ws_size,
                              hipStream_t stream) {
  (void)in_sizes; (void)n_in; (void)out_size; (void)ws_size;
  const float* nodes     = (const float*)d_in[0];
  const float* edges     = (const float*)d_in[1];
  const int*   senders   = (const int*)d_in[2];
  const int*   receivers = (const int*)d_in[3];
  const float* W_node    = (const float*)d_in[4];
  const float* W_rel     = (const float*)d_in[5];
  const float* ln_scale  = (const float*)d_in[6];
  const float* ln_bias   = (const float*)d_in[7];
  float* out = (float*)d_out;
  int*   deg = (int*)d_ws;  // R*N ints, then reused in place as inverse-degree floats

  zero_kernel  <<<2048, 256, 0, stream>>>(out, deg);
  degree_kernel<<<2048, 256, 0, stream>>>(receivers, deg);
  invdeg_kernel<<<1024, 256, 0, stream>>>(deg);
  edge_gemm_kernel<<<N_REL * BPR, 256, 0, stream>>>(nodes, edges, senders, receivers,
                                                    W_rel, (const float*)deg, out);
  final_kernel<<<N_NODES / TM2, 256, 0, stream>>>(nodes, W_node, ln_scale, ln_bias, out);
}
